// ZuoErJinYouErChu_26079041421872
// MI455X (gfx1250) — matrix operand order repaired and checked by reading, not compiled, not run on hardware
//
#include <hip/hip_runtime.h>
#include <hip/hip_bf16.h>

// ---------------- problem constants ----------------
constexpr int kBd = 8, kSd = 4096, kDd = 1024, kMd = 64;
constexpr int kNtok = kBd * kSd;          // 32768 tokens
constexpr int kCH = 16;                   // scan chunks per sequence
constexpr int kL  = kSd / kCH;            // 256 steps per chunk

// ---------------- CDNA5 WMMA types ----------------
typedef __attribute__((ext_vector_type(16))) __bf16 v16bf;
typedef __attribute__((ext_vector_type(8)))  float  v8f;
typedef __attribute__((ext_vector_type(4)))  unsigned int v4u;   // builtin 16B chunk

#define LDS_AS __attribute__((address_space(3)))

using bf16 = __hip_bfloat16;

// ---------------- fragment loaders, GLOBAL source (ISA 7.12.2) ----------
// A-matrix 16x32 bf16, row-major [row][K]:
//   lane<16 : row=lane,    K = {k0..k0+7} and {k0+16..k0+23}
//   lane>=16: row=lane-16, K = {k0+8..k0+15} and {k0+24..k0+31}
__device__ __forceinline__ v16bf load_fragA(const bf16* base, int ld, int row0,
                                            int k0, int lane) {
    int r  = lane & 15;
    int kl = (lane >> 4) << 3;                 // 0 or 8
    const bf16* p = base + (size_t)(row0 + r) * ld + k0 + kl;
    union { v16bf v; v4u q[2]; } u;
    u.q[0] = *(const v4u*)(p);                 // K = kl .. kl+7
    u.q[1] = *(const v4u*)(p + 16);            // K = kl+16 .. kl+23
    return u.v;
}

// B-matrix 32x16 bf16, stored transposed [col][K] (per ISA B layout:
// lane<16 holds K=0..7,16..23 of column lane; lane>=16 holds K=8..15,24..31)
__device__ __forceinline__ v16bf load_fragB(const bf16* baseT, int ld, int col0,
                                            int k0, int lane) {
    int n  = lane & 15;
    int kb = (lane >> 4) << 3;                 // 0 or 8
    const bf16* p = baseT + (size_t)(col0 + n) * ld + k0 + kb;
    union { v16bf v; v4u q[2]; } u;
    u.q[0] = *(const v4u*)(p);                 // K = kb .. kb+7
    u.q[1] = *(const v4u*)(p + 16);            // K = kb+16 .. kb+23
    return u.v;
}

// ---------------- fragment loaders, LDS source (explicit as3 -> ds_load) -
// off/base are LDS byte offsets; ldB is the padded row stride in bytes.
__device__ __forceinline__ v16bf load_fragA_lds(unsigned base, int ldB,
                                                int row0, int k0, int lane) {
    int r  = lane & 15;
    int kl = (lane >> 4) << 3;                 // 0 or 8
    unsigned off = base + (unsigned)(row0 + r) * (unsigned)ldB
                        + (unsigned)(k0 + kl) * 2u;
    union { v16bf v; v4u q[2]; } u;
    u.q[0] = *(const LDS_AS v4u*)(uintptr_t)(off);          // K = kl .. kl+7
    u.q[1] = *(const LDS_AS v4u*)(uintptr_t)(off + 32u);    // K = kl+16 .. kl+23
    return u.v;
}

__device__ __forceinline__ v16bf load_fragB_lds(unsigned base, int ldB,
                                                int col0, int k0, int lane) {
    int n  = lane & 15;
    int kb = (lane >> 4) << 3;                 // 0 or 8
    unsigned off = base + (unsigned)(col0 + n) * (unsigned)ldB
                        + (unsigned)(k0 + kb) * 2u;
    union { v16bf v; v4u q[2]; } u;
    u.q[0] = *(const LDS_AS v4u*)(uintptr_t)(off);          // K = kb .. kb+7
    u.q[1] = *(const LDS_AS v4u*)(uintptr_t)(off + 32u);    // K = kb+16 .. kb+23
    return u.v;
}

__device__ __forceinline__ v8f wmma_bf16(v16bf a, v16bf b, v8f c) {
    return __builtin_amdgcn_wmma_f32_16x16x32_bf16(
        /*neg_a=*/false, a, /*neg_b=*/false, b,
        /*c_mod=*/(short)0, c, /*reuse_a=*/false, /*reuse_b=*/false);
}

__device__ __forceinline__ float sigmoidf_(float x) {
    return 1.0f / (1.0f + __expf(-x));
}

// --------- CDNA5 async global->LDS copy (ASYNCcnt-tracked, ISA ch.10) ---
// 16 bytes per lane: LDS byte offset in a VGPR, 64-bit global addr in VGPR pair.
__device__ __forceinline__ void async_copy_b128(unsigned lds_off, const void* gaddr) {
    asm volatile("global_load_async_to_lds_b128 %0, %1, off"
                 :: "v"(lds_off), "v"(gaddr)
                 : "memory");
}
__device__ __forceinline__ void wait_asynccnt_6() {
    asm volatile("s_wait_asynccnt 0x6" ::: "memory");
}
__device__ __forceinline__ void wait_asynccnt_0() {
    asm volatile("s_wait_asynccnt 0x0" ::: "memory");
}

// ---------------- 1) convert x (f32 -> bf16), [Ntok][D] ----------------
__global__ void k_convert_x(const float* __restrict__ x, bf16* __restrict__ xb,
                            int n4) {
    int g = blockIdx.x * blockDim.x + threadIdx.x;
    if (g >= n4) return;
    const float4 v = ((const float4*)x)[g];
    bf16* o = xb + (size_t)g * 4;
    o[0] = __float2bfloat16(v.x);
    o[1] = __float2bfloat16(v.y);
    o[2] = __float2bfloat16(v.z);
    o[3] = __float2bfloat16(v.w);
}

// ---------------- 2) pack [Ww|Wr|Wt] -> bf16 transposed [192][1024] -----
__global__ void k_pack_wwrt(const float* __restrict__ Ww,
                            const float* __restrict__ Wr,
                            const float* __restrict__ Wt,
                            bf16* __restrict__ dstT) {
    int g = blockIdx.x * blockDim.x + threadIdx.x;   // g = n*1024 + k
    if (g >= 192 * kDd) return;
    int n = g / kDd, k = g % kDd;
    float v;
    if (n < 64)       v = Ww[(size_t)k * kMd + n];
    else if (n < 128) v = Wr[(size_t)k * kMd + (n - 64)];
    else              v = Wt[(size_t)k * kMd + (n - 128)];
    dstT[g] = __float2bfloat16(v);
}

// ---------------- 3) pack WoTop -> bf16 transposed [1024][1024] ---------
__global__ void k_pack_wotop(const float* __restrict__ Wo, bf16* __restrict__ dstT) {
    int g = blockIdx.x * blockDim.x + threadIdx.x;   // g = n*1024 + k
    if (g >= kDd * kDd) return;
    int n = g / kDd, k = g % kDd;
    dstT[g] = __float2bfloat16(Wo[(size_t)k * kDd + n]);   // WoTop[k][n]
}

// ---------------- 4) W2 = Wp @ WoBot, stored transposed bf16 [1024][64] --
__global__ void k_compute_w2(const float* __restrict__ Wp,
                             const float* __restrict__ Wo,
                             bf16* __restrict__ w2T) {
    int j = blockIdx.x * blockDim.x + threadIdx.x;   // output column 0..1023
    int m = blockIdx.y;                              // 0..63
    float acc = 0.0f;
    const float* wob = Wo + (size_t)kDd * kDd;       // WoBot base
    for (int k = 0; k < kDd; ++k)
        acc += Wp[(size_t)m * kDd + k] * wob[(size_t)k * kDd + j];
    w2T[(size_t)j * kMd + m] = __float2bfloat16(acc);
}

// ---------------- 5) bias2 = bp @ WoBot + bo, f32 [1024] ----------------
__global__ void k_compute_bias2(const float* __restrict__ bp,
                                const float* __restrict__ Wo,
                                const float* __restrict__ bo,
                                float* __restrict__ bias2) {
    int j = blockIdx.x * blockDim.x + threadIdx.x;
    const float* wob = Wo + (size_t)kDd * kDd;
    float acc = bo[j];
    for (int k = 0; k < kDd; ++k)
        acc += bp[k] * wob[(size_t)k * kDd + j];
    bias2[j] = acc;
}

// ---------------- 6) projection GEMM: [32768,1024]x[1024,192] -----------
// block = 256 thr = 8 waves (4x2), block tile 128 rows x 64 cols
__global__ void k_proj_gemm(const bf16* __restrict__ X,      // [Ntok][1024]
                            const bf16* __restrict__ WT,     // [192][1024]
                            const float* __restrict__ bw,
                            const float* __restrict__ br,
                            const float* __restrict__ bt,
                            float* __restrict__ ww,          // [Ntok][64]
                            float* __restrict__ rg,
                            float* __restrict__ mt) {
    const int lane = threadIdx.x & 31;
    const int wid  = threadIdx.x >> 5;
    const int rowBase = blockIdx.x * 128 + (wid >> 1) * 32;
    const int colBase = blockIdx.y * 64  + (wid & 1) * 32;

    v8f acc[2][2];
    #pragma unroll
    for (int i = 0; i < 2; ++i)
        #pragma unroll
        for (int j = 0; j < 2; ++j)
            acc[i][j] = v8f{0.f,0.f,0.f,0.f,0.f,0.f,0.f,0.f};

    for (int k0 = 0; k0 < kDd; k0 += 32) {
        v16bf a0 = load_fragA(X, kDd, rowBase,      k0, lane);
        v16bf a1 = load_fragA(X, kDd, rowBase + 16, k0, lane);
        v16bf b0 = load_fragB(WT, kDd, colBase,      k0, lane);
        v16bf b1 = load_fragB(WT, kDd, colBase + 16, k0, lane);
        acc[0][0] = wmma_bf16(a0, b0, acc[0][0]);
        acc[0][1] = wmma_bf16(a0, b1, acc[0][1]);
        acc[1][0] = wmma_bf16(a1, b0, acc[1][0]);
        acc[1][1] = wmma_bf16(a1, b1, acc[1][1]);
    }

    const int rsub = (lane >> 4) << 3;   // 0 or 8
    const int csub = lane & 15;
    #pragma unroll
    for (int i = 0; i < 2; ++i) {
        #pragma unroll
        for (int j = 0; j < 2; ++j) {
            int col = colBase + j * 16 + csub;   // global n in [0,192)
            #pragma unroll
            for (int r = 0; r < 8; ++r) {
                int row = rowBase + i * 16 + rsub + r;   // token index
                float v = acc[i][j][r];
                size_t o;
                if (col < 64) {
                    o = (size_t)row * kMd + col;
                    ww[o] = sigmoidf_(v + bw[col]);
                } else if (col < 128) {
                    o = (size_t)row * kMd + (col - 64);
                    rg[o] = sigmoidf_(v + br[col - 64]);
                } else {
                    o = (size_t)row * kMd + (col - 128);
                    mt[o] = v + bt[col - 128];
                }
            }
        }
    }
}

// ---------------- 7) scan pass 1: per-chunk affine composites -----------
__global__ void k_scan_pass1(const float* __restrict__ ww,
                             const float* __restrict__ mt,
                             float* __restrict__ Ac, float* __restrict__ Bc) {
    int b = blockIdx.x / kCH, c = blockIdx.x % kCH, m = threadIdx.x;
    size_t base = ((size_t)b * kSd + (size_t)c * kL) * kMd + m;
    float A = 1.0f, Bv = 0.0f;
    for (int t = 0; t < kL; ++t) {
        float w  = ww[base + (size_t)t * kMd];
        float mm = mt[base + (size_t)t * kMd];
        float a  = 1.0f - w;
        Bv = a * Bv + w * mm;
        A  = a * A;
    }
    size_t idx = ((size_t)(b * kMd + m)) * kCH + c;
    Ac[idx] = A;
    Bc[idx] = Bv;
}

// ---------------- 8) scan pass 2: prefix over chunks --------------------
__global__ void k_scan_pass2(const float* __restrict__ Ac,
                             const float* __restrict__ Bc,
                             float* __restrict__ Sinit) {
    int t = threadIdx.x;     // 0..511
    float s = 0.0f;
    size_t base = (size_t)t * kCH;
    for (int c = 0; c < kCH; ++c) {
        Sinit[base + c] = s;
        s = Ac[base + c] * s + Bc[base + c];
    }
}

// ---------------- 9) scan pass 3: emit rs = rg * s_prev (bf16) ----------
__global__ void k_scan_pass3(const float* __restrict__ ww,
                             const float* __restrict__ mt,
                             const float* __restrict__ rg,
                             const float* __restrict__ Sinit,
                             bf16* __restrict__ rsb) {
    int b = blockIdx.x / kCH, c = blockIdx.x % kCH, m = threadIdx.x;
    size_t base = ((size_t)b * kSd + (size_t)c * kL) * kMd + m;
    float s = Sinit[((size_t)(b * kMd + m)) * kCH + c];
    for (int t = 0; t < kL; ++t) {
        size_t idx = base + (size_t)t * kMd;
        float w  = ww[idx];
        float mm = mt[idx];
        float r  = rg[idx];
        rsb[idx] = __float2bfloat16(r * s);     // uses PRE-update state
        s = (1.0f - w) * s + w * mm;
    }
}

// ---------------- 10) fused final GEMM + tanh ---------------------------
// out = tanh( X@WoTopT  +  RS@W2T  + bias2 )   [32768 x 1024]
// block = 256 thr = 8 waves (2x4), block tile 64 rows x 128 cols.
// Phase 1 (K=1024): double-buffered LDS pipeline, K=64 per staged tile,
// filled with GLOBAL_LOAD_ASYNC_TO_LDS_B128 (ASYNCcnt), fragments read
// back via explicit addrspace(3) ds_load_b128.
constexpr int kKT     = 64;                     // K per staged tile
constexpr int kRowB   = (kKT + 8) * 2;          // 144 B padded row stride
constexpr int kABytes = 64  * kRowB;            // 9216 B  (A tile 64x64)
constexpr int kBBytes = 128 * kRowB;            // 18432 B (B tile 128x64)
constexpr int kBufB   = kABytes + kBBytes;      // 27648 B per buffer

__global__ void k_final_gemm(const bf16* __restrict__ X,     // [Ntok][1024]
                             const bf16* __restrict__ WoT,   // [1024][1024]
                             const bf16* __restrict__ RS,    // [Ntok][64]
                             const bf16* __restrict__ W2T,   // [1024][64]
                             const float* __restrict__ bias2,
                             float* __restrict__ out) {
    __shared__ __align__(16) char smem[2 * kBufB];
    // Low 32 bits of a generic LDS address are the LDS byte offset
    // (ISA 10.2 aperture rule) -> usable by both the async asm dst and
    // the as3 ds_load pointers.
    const unsigned smemBase = (unsigned)(uintptr_t)(void*)smem;

    const int lane = threadIdx.x & 31;
    const int wid  = threadIdx.x >> 5;
    const int wr   = wid >> 2;                  // 0..1 (row tile of 32)
    const int wc   = wid & 3;                   // 0..3 (col tile of 32)
    const int blockRow = blockIdx.x * 64;
    const int blockCol = blockIdx.y * 128;

    v8f acc[2][2];
    #pragma unroll
    for (int i = 0; i < 2; ++i)
        #pragma unroll
        for (int j = 0; j < 2; ++j)
            acc[i][j] = v8f{0.f,0.f,0.f,0.f,0.f,0.f,0.f,0.f};

    const int t = threadIdx.x;
    // per-thread fill of one K=64 tile: 2 A chunks + 4 B chunks (6 asyncs)
    auto fill_tile = [&](int buf, int k0) {
        unsigned base = smemBase + (unsigned)(buf * kBufB);
        #pragma unroll
        for (int h = 0; h < 2; ++h) {   // A tile: 64 rows x 64 K, 16B chunks
            int c = t + h * 256;
            int arow = c >> 3, akc = c & 7;
            unsigned lds = base + arow * kRowB + akc * 16;
            const bf16* g = X + (size_t)(blockRow + arow) * kDd + k0 + akc * 8;
            async_copy_b128(lds, g);
        }
        #pragma unroll
        for (int h = 0; h < 4; ++h) {   // B tile: 128 cols x 64 K
            int c = t + h * 256;
            int col = c >> 3, bkc = c & 7;
            unsigned lds = base + kABytes + col * kRowB + bkc * 16;
            const bf16* g = WoT + (size_t)(blockCol + col) * kDd + k0 + bkc * 8;
            async_copy_b128(lds, g);
        }
    };

    // phase 1: K = 1024 over x, double-buffered async LDS pipeline
    fill_tile(0, 0);
    for (int k0 = 0; k0 < kDd; k0 += kKT) {
        const int cur = (k0 / kKT) & 1;
        const bool hasNext = (k0 + kKT) < kDd;
        if (hasNext) {
            fill_tile(cur ^ 1, k0 + kKT);
            wait_asynccnt_6();          // retire current tile's 6 asyncs only
        } else {
            wait_asynccnt_0();
        }
        __syncthreads();                // cross-wave LDS visibility

        const unsigned aBase = smemBase + (unsigned)(cur * kBufB);
        const unsigned bBase = aBase + kABytes;
        #pragma unroll
        for (int kk = 0; kk < kKT; kk += 32) {
            v16bf a0 = load_fragA_lds(aBase, kRowB, wr * 32,      kk, lane);
            v16bf a1 = load_fragA_lds(aBase, kRowB, wr * 32 + 16, kk, lane);
            v16bf b0 = load_fragB_lds(bBase, kRowB, wc * 32,      kk, lane);
            v16bf b1 = load_fragB_lds(bBase, kRowB, wc * 32 + 16, kk, lane);
            acc[0][0] = wmma_bf16(a0, b0, acc[0][0]);
            acc[0][1] = wmma_bf16(a0, b1, acc[0][1]);
            acc[1][0] = wmma_bf16(a1, b0, acc[1][0]);
            acc[1][1] = wmma_bf16(a1, b1, acc[1][1]);
        }
        __syncthreads();                // reads done before buffer reuse
    }

    // phase 2: K = 64 over rs (2 steps, direct global)
    const int rowBase = blockRow + wr * 32;
    const int colBase = blockCol + wc * 32;
    #pragma unroll
    for (int k0 = 0; k0 < kMd; k0 += 32) {
        v16bf a0 = load_fragA(RS, kMd, rowBase,      k0, lane);
        v16bf a1 = load_fragA(RS, kMd, rowBase + 16, k0, lane);
        v16bf b0 = load_fragB(W2T, kMd, colBase,      k0, lane);
        v16bf b1 = load_fragB(W2T, kMd, colBase + 16, k0, lane);
        acc[0][0] = wmma_bf16(a0, b0, acc[0][0]);
        acc[0][1] = wmma_bf16(a0, b1, acc[0][1]);
        acc[1][0] = wmma_bf16(a1, b0, acc[1][0]);
        acc[1][1] = wmma_bf16(a1, b1, acc[1][1]);
    }

    const int rsub = (lane >> 4) << 3;
    const int csub = lane & 15;
    #pragma unroll
    for (int i = 0; i < 2; ++i) {
        #pragma unroll
        for (int j = 0; j < 2; ++j) {
            int col = colBase + j * 16 + csub;
            float b2 = bias2[col];
            #pragma unroll
            for (int r = 0; r < 8; ++r) {
                int row = rowBase + i * 16 + rsub + r;
                out[(size_t)row * kDd + col] = tanhf(acc[i][j][r] + b2);
            }
        }
    }
}

// ---------------- host launcher ----------------
extern "C" void kernel_launch(void* const* d_in, const int* in_sizes, int n_in,
                              void* d_out, int out_size, void* d_ws, size_t ws_size,
                              hipStream_t stream) {
    const float* x  = (const float*)d_in[0];
    const float* Ww = (const float*)d_in[1];
    const float* bw = (const float*)d_in[2];
    const float* Wr = (const float*)d_in[3];
    const float* br = (const float*)d_in[4];
    const float* Wt = (const float*)d_in[5];
    const float* bt = (const float*)d_in[6];
    const float* Wp = (const float*)d_in[7];
    const float* bp = (const float*)d_in[8];
    const float* Wo = (const float*)d_in[9];
    const float* bo = (const float*)d_in[10];
    float* out = (float*)d_out;

    // workspace layout (bytes)
    char* ws = (char*)d_ws;
    size_t o = 0;
    auto take = [&](size_t bytes) { char* p = ws + o; o = (o + bytes + 255) & ~(size_t)255; return p; };
    bf16*  xbf    = (bf16*)take((size_t)kNtok * kDd * sizeof(bf16));     // 64 MiB
    bf16*  wwrtT  = (bf16*)take((size_t)192 * kDd * sizeof(bf16));
    bf16*  woTopT = (bf16*)take((size_t)kDd * kDd * sizeof(bf16));
    bf16*  w2T    = (bf16*)take((size_t)kDd * kMd * sizeof(bf16));
    float* bias2  = (float*)take((size_t)kDd * sizeof(float));
    float* wwA    = (float*)take((size_t)kNtok * kMd * sizeof(float));
    float* rgA    = (float*)take((size_t)kNtok * kMd * sizeof(float));
    float* mtA    = (float*)take((size_t)kNtok * kMd * sizeof(float));
    float* Ac     = (float*)take((size_t)kBd * kMd * kCH * sizeof(float));
    float* Bc     = (float*)take((size_t)kBd * kMd * kCH * sizeof(float));
    float* Sinit  = (float*)take((size_t)kBd * kMd * kCH * sizeof(float));
    bf16*  rsbf   = (bf16*)take((size_t)kNtok * kMd * sizeof(bf16));
    (void)ws_size; (void)in_sizes; (void)n_in; (void)out_size;

    // 1) x -> bf16
    {
        int n4 = (kNtok * kDd) / 4;
        k_convert_x<<<dim3(n4 / 256), dim3(256), 0, stream>>>(x, xbf, n4);
    }
    // 2/3) weight packing
    k_pack_wwrt<<<dim3((192 * kDd) / 256), dim3(256), 0, stream>>>(Ww, Wr, Wt, wwrtT);
    k_pack_wotop<<<dim3((kDd * kDd) / 256), dim3(256), 0, stream>>>(Wo, woTopT);
    // 4/5) folded readout weight + bias
    k_compute_w2<<<dim3(kDd / 256, kMd), dim3(256), 0, stream>>>(Wp, Wo, w2T);
    k_compute_bias2<<<dim3(kDd / 256), dim3(256), 0, stream>>>(bp, Wo, bo, bias2);
    // 6) projections (WMMA)
    k_proj_gemm<<<dim3(kNtok / 128, 3), dim3(256), 0, stream>>>(
        xbf, wwrtT, bw, br, bt, wwA, rgA, mtA);
    // 7/8/9) chunked affine scan
    k_scan_pass1<<<dim3(kBd * kCH), dim3(kMd), 0, stream>>>(wwA, mtA, Ac, Bc);
    k_scan_pass2<<<dim3(1), dim3(kBd * kMd), 0, stream>>>(Ac, Bc, Sinit);
    k_scan_pass3<<<dim3(kBd * kCH), dim3(kMd), 0, stream>>>(wwA, mtA, rgA, Sinit, rsbf);
    // 10) fused final GEMM + tanh (WMMA + async-LDS pipeline)
    k_final_gemm<<<dim3(kNtok / 64, kDd / 128), dim3(256), 0, stream>>>(
        xbf, woTopT, rsbf, w2T, bias2, out);
}
